// GridNetBlock_1_43739946943059
// MI455X (gfx1250) — matrix operand order repaired and checked by reading, not compiled, not run on hardware
//
#include <hip/hip_runtime.h>
#include <hip/hip_bf16.h>
#include <math.h>

typedef __bf16 bf16;
typedef __attribute__((ext_vector_type(16))) __bf16 v16bf;
typedef __attribute__((ext_vector_type(8)))  __bf16 v8bf;
typedef __attribute__((ext_vector_type(2)))  __bf16 v2bf;
typedef __attribute__((ext_vector_type(8)))  float  v8f;

#define LN_EPS 1e-5f

// ---------------------------------------------------------------------------
// WMMA fragment helpers for v_wmma_f32_16x16x32_bf16 (wave32).
// A (16x32, 16-bit):  lane[3:0]=M row, lane[4]=K-half(0/8); VGPR i<4 -> K=kh+2i,
//                     VGPR i>=4 -> K=16+kh+2(i-4). Per lane this is two
//                     contiguous 8-element (16B) runs: K=[kh,kh+8) and
//                     K=[16+kh,16+kh+8).                    (ISA 7.12.2)
// B (32x16, 16-bit):  lane[3:0]=N col, lane[4]=K-half(0/8); same K order as A.
//                     With a K-contiguous (column-major) staging buffer this is
//                     two contiguous 8-element (16B) runs per lane.
// C/D (16x16, f32):   VGPR r: lanes 0-15 -> M=r, lanes 16-31 -> M=r+8; N=lane&15.
// ---------------------------------------------------------------------------
__device__ __forceinline__ v8f zero8() {
  v8f z = {0.f, 0.f, 0.f, 0.f, 0.f, 0.f, 0.f, 0.f};
  return z;
}

__device__ __forceinline__ v8f wmma_bf16(v16bf a, v16bf b, v8f c) {
  return __builtin_amdgcn_wmma_f32_16x16x32_bf16(false, a, false, b, (short)0, c,
                                                 false, false);
}

__device__ __forceinline__ v16bf combine16(v8bf lo, v8bf hi) {
  v16bf r;
#pragma unroll
  for (int i = 0; i < 8; ++i) { r[i] = lo[i]; r[i + 8] = hi[i]; }
  return r;
}

// A fragment from row-major [M x K] buffer (element (m,k) at p[m*ld + k]).
// ld elements must be a multiple of 8 (16B rows); two b128 loads per lane.
__device__ __forceinline__ v16bf frag_a(const bf16* p, int m0, int k0, int ld,
                                        int lane) {
  int row = m0 + (lane & 15);
  int kh  = (lane >> 4) << 3;  // 0 or 8
  const bf16* q = p + (size_t)row * ld + k0 + kh;
  return combine16(*(const v8bf*)q, *(const v8bf*)(q + 16));
}

// B fragment from K-contiguous staging (element (k,n) at p[n*ldk + k]).
// Two 16B runs per lane (K=[kh,kh+8) and K=[16+kh,16+kh+8)) -> two b128 loads.
__device__ __forceinline__ v16bf frag_bc(const bf16* p, int k0, int n0, int ldk,
                                         int lane) {
  int col = n0 + (lane & 15);
  int kg  = k0 + ((lane >> 4) << 3);  // 0 or 8
  const bf16* q = p + (size_t)col * ldk + kg;
  return combine16(*(const v8bf*)q, *(const v8bf*)(q + 16));
}

__device__ __forceinline__ v2bf pack2(float a, float b) {
  v2bf r;
  r[0] = (bf16)a;
  r[1] = (bf16)b;
  return r;
}

// ---------------------------------------------------------------------------
// Generic WMMA GEMM:  C[M x N] = scale * (A[M x K](bf16) @ Bw[K x N](f32))
// Weights converted to bf16, K-contiguous, in LDS.  4 waves per block; each
// wave owns a 16x32 output tile (two WMMA accumulators sharing one A frag).
// M % 64 == 0, N % 32 == 0, K % 32 == 0, K <= 256.
// ---------------------------------------------------------------------------
template <bool OUT_BF16>
__global__ void __launch_bounds__(128)
gemm_wmma_kernel(const bf16* __restrict__ A, const float* __restrict__ Bw,
                 float* __restrict__ Cf, bf16* __restrict__ Cb,
                 int M, int N, int K, float scale) {
  __shared__ __align__(16) bf16 sB[32 * 256];  // [col][K]
  const int n0  = blockIdx.x * 32;
  const int tid = threadIdx.x;
  for (int idx = tid; idx < 32 * K; idx += 128) {
    int nn = idx / K, kk = idx - nn * K;
    sB[idx] = (bf16)Bw[(size_t)kk * N + n0 + nn];
  }
  __syncthreads();

  const int wave = tid >> 5, lane = tid & 31;
  const int m0 = blockIdx.y * 64 + wave * 16;

  v8f acc0 = zero8(), acc1 = zero8();
  const bf16* Arow = A + (size_t)m0 * K;
  for (int k0 = 0; k0 < K; k0 += 32) {
    __builtin_prefetch(Arow + 16 * (size_t)K + k0, 0, 1);  // global_prefetch_b8
    v16bf a  = frag_a(Arow, 0, k0, K, lane);
    v16bf b0 = frag_bc(sB, k0, 0, K, lane);
    v16bf b1 = frag_bc(sB + 16 * (size_t)K, k0, 0, K, lane);
    acc0 = wmma_bf16(a, b0, acc0);
    acc1 = wmma_bf16(a, b1, acc1);
  }

  const int mrow = m0 + ((lane >> 4) << 3);
  const int col  = n0 + (lane & 15);
#pragma unroll
  for (int r = 0; r < 8; ++r) {
    size_t o = (size_t)(mrow + r) * N + col;
    float v0 = acc0[r] * scale;
    float v1 = acc1[r] * scale;
    if (OUT_BF16) {
      Cb[o]      = (bf16)v0;
      Cb[o + 16] = (bf16)v1;
    } else {
      Cf[o]      = v0;
      Cf[o + 16] = v1;
    }
  }
}

// ---------------------------------------------------------------------------
// Channel LayerNorm over C=64 of x[B,C,T,Q]; one thread per (b,t,q).
// ---------------------------------------------------------------------------
__global__ void ln4d_kernel(const float* __restrict__ x,
                            const float* __restrict__ gam,
                            const float* __restrict__ bet,
                            float* __restrict__ u, int Cc, int TQ, int total) {
  int idx = blockIdx.x * blockDim.x + threadIdx.x;
  if (idx >= total) return;
  int b = idx / TQ, tq = idx - b * TQ;
  const float* px = x + (size_t)b * Cc * TQ + tq;
  float s1 = 0.f, s2 = 0.f;
  for (int c = 0; c < Cc; ++c) {
    float v = px[(size_t)c * TQ];
    s1 += v;
    s2 += v * v;
  }
  float mu  = s1 / (float)Cc;
  float var = s2 / (float)Cc - mu * mu;
  float inv = rsqrtf(var + LN_EPS);
  float* pu = u + (size_t)b * Cc * TQ + tq;
  for (int c = 0; c < Cc; ++c)
    pu[(size_t)c * TQ] = (px[(size_t)c * TQ] - mu) * inv * gam[c] + bet[c];
}

// ---------------------------------------------------------------------------
// Unfold + pack to bf16 GEMM A-matrix: fA[(n*Lp + l), 2c+k] = u_seq[n, c, l+k]
// ---------------------------------------------------------------------------
__global__ void unfold_kernel(const float* __restrict__ u, bf16* __restrict__ fA,
                              int Lp, int L, int inner, int strB, int strC,
                              int strI, int strP, long total) {
  long idx = blockIdx.x * (long)blockDim.x + threadIdx.x;
  if (idx >= total) return;
  int col = (int)(idx & 127);
  long rl = idx >> 7;
  int l = (int)(rl % Lp);
  int n = (int)(rl / Lp);
  float val = 0.f;
  if (l < L) {
    int c = col >> 1, k = col & 1;
    int bI = n / inner, iI = n - bI * inner;
    val = u[(size_t)bI * strB + (size_t)c * strC + (size_t)iI * strI +
            (size_t)(l + k) * strP];
  }
  fA[idx] = (bf16)val;
}

// gk = log_sigmoid(t)/32, numerically stable
__global__ void logsig_kernel(float* __restrict__ gk, long total) {
  long idx = blockIdx.x * (long)blockDim.x + threadIdx.x;
  if (idx >= total) return;
  float t = gk[idx];
  gk[idx] = (fminf(t, 0.f) - log1pf(expf(-fabsf(t)))) * (1.f / 32.f);
}

// ---------------------------------------------------------------------------
// Chunked GLA: one wave32 per (sequence n, head h).  The 64x64 recurrent state
// S lives in 16 WMMA accumulator fragments (128 VGPRs) across all chunks.
// Per chunk (CS=32):  gc=cumsum(gk);  A=(q e^gc)(k e^-gc)^T masked;
// o = A v + (q e^gc) S;  S = e^gl * S + (k e^(gl-gc))^T v.
// All operand tiles staged bf16 in LDS with K-contiguous layouts so every
// fragment load is two ds_load_b128.
// ---------------------------------------------------------------------------
__global__ void __launch_bounds__(32)
gla_kernel(const float* __restrict__ qG, const float* __restrict__ kG,
           const float* __restrict__ vG, const float* __restrict__ gkG,
           float* __restrict__ oG, int nc) {
  __shared__ __align__(16) float s_gk[32 * 64];
  __shared__ __align__(16) float s_gc[32 * 64];
  __shared__ __align__(16) bf16  s_qi [32 * 64];  // [pos][dim]  (A operand)
  __shared__ __align__(16) bf16  s_ki [32 * 64];  // [pos][dim]  (B^T: k-contig)
  __shared__ __align__(16) bf16  s_koT[64 * 32];  // [dim][pos]  (A of k_o^T)
  __shared__ __align__(16) bf16  s_vT [64 * 32];  // [dim][pos]  (B: k-contig)
  __shared__ __align__(16) bf16  s_Am [32 * 32];  // [pos][pos]  (A operand)
  __shared__ __align__(16) bf16  s_SbT[64 * 64];  // [dv][dk]    (B: k-contig)

  const int lane = threadIdx.x;
  const int h = blockIdx.x & 3;  // H = 4
  const int n = blockIdx.x >> 2;
  const size_t rowbase0 = (size_t)n * (size_t)(nc * 32);
  const int colb = h * 64;
  const int hi8 = (lane >> 4) << 3;
  const int lx  = lane & 15;

  v8f S[4][4];
#pragma unroll
  for (int i = 0; i < 4; ++i)
#pragma unroll
    for (int j = 0; j < 4; ++j) S[i][j] = zero8();

  for (int c = 0; c < nc; ++c) {
    const size_t rb = rowbase0 + (size_t)c * 32;
    const size_t gbase = (rb + lane) * 256 + colb;

    // ---- stage gk chunk (lane = position), float4 loads ----
    {
      const float4* pg = (const float4*)(gkG + gbase);
      float4* sg = (float4*)(s_gk + lane * 64);
#pragma unroll
      for (int i = 0; i < 16; ++i) sg[i] = pg[i];
    }
    __syncthreads();

    // ---- cumulative sum over positions (each lane owns 2 dims) ----
    {
      int d0 = lane * 2;
      float r0 = 0.f, r1 = 0.f;
      for (int t = 0; t < 32; ++t) {
        r0 += s_gk[t * 64 + d0];     s_gc[t * 64 + d0]     = r0;
        r1 += s_gk[t * 64 + d0 + 1]; s_gc[t * 64 + d0 + 1] = r1;
      }
    }
    __syncthreads();

    // ---- build bf16 tiles q_i, k_i (row-major) and k_o^T, v^T (transposed) ----
    {
      const float4* pq = (const float4*)(qG + gbase);  // q pre-scaled by Dk^-.5
      const float4* pk = (const float4*)(kG + gbase);
      const float4* pv = (const float4*)(vG + gbase);
      const float4* pgc = (const float4*)(s_gc + lane * 64);
      const float4* pgl = (const float4*)(s_gc + 31 * 64);
#pragma unroll
      for (int i = 0; i < 16; ++i) {
        float4 g4 = pgc[i], l4 = pgl[i], q4 = pq[i], k4 = pk[i], v4 = pv[i];
        int d = 4 * i;
        float e0 = expf(g4.x), e1 = expf(g4.y), e2 = expf(g4.z), e3 = expf(g4.w);
        // q_i = q * e^gc
        *(v2bf*)(s_qi + lane * 64 + d)     = pack2(q4.x * e0, q4.y * e1);
        *(v2bf*)(s_qi + lane * 64 + d + 2) = pack2(q4.z * e2, q4.w * e3);
        // k_i = k * e^-gc
        *(v2bf*)(s_ki + lane * 64 + d)     = pack2(k4.x / e0, k4.y / e1);
        *(v2bf*)(s_ki + lane * 64 + d + 2) = pack2(k4.z / e2, k4.w / e3);
        // k_o = k * e^(gl-gc)   (transposed store: [dim][pos])
        s_koT[(d + 0) * 32 + lane] = (bf16)(k4.x * expf(l4.x - g4.x));
        s_koT[(d + 1) * 32 + lane] = (bf16)(k4.y * expf(l4.y - g4.y));
        s_koT[(d + 2) * 32 + lane] = (bf16)(k4.z * expf(l4.z - g4.z));
        s_koT[(d + 3) * 32 + lane] = (bf16)(k4.w * expf(l4.w - g4.w));
        // v (transposed store: [dim][pos])
        s_vT[(d + 0) * 32 + lane] = (bf16)v4.x;
        s_vT[(d + 1) * 32 + lane] = (bf16)v4.y;
        s_vT[(d + 2) * 32 + lane] = (bf16)v4.z;
        s_vT[(d + 3) * 32 + lane] = (bf16)v4.w;
      }
    }

    // ---- dump pre-update state S transposed: s_SbT[dv][dk] (k-contig B) ----
#pragma unroll
    for (int mt = 0; mt < 4; ++mt)
#pragma unroll
      for (int nt = 0; nt < 4; ++nt)
#pragma unroll
        for (int r = 0; r < 8; r += 2)
          *(v2bf*)(s_SbT + (nt * 16 + lx) * 64 + mt * 16 + hi8 + r) =
              pack2(S[mt][nt][r], S[mt][nt][r + 1]);
    __syncthreads();

    // ---- A = q_i @ k_i^T  (32x32), causal mask, store bf16 ----
#pragma unroll
    for (int mt = 0; mt < 2; ++mt)
#pragma unroll
      for (int nt = 0; nt < 2; ++nt) {
        v8f acc = zero8();
#pragma unroll
        for (int kc = 0; kc < 2; ++kc) {
          v16bf a = frag_a(s_qi, mt * 16, kc * 32, 64, lane);
          v16bf b = frag_bc(s_ki, kc * 32, nt * 16, 64, lane);  // (k,n)->s_ki[n][k]
          acc = wmma_bf16(a, b, acc);
        }
#pragma unroll
        for (int r = 0; r < 8; ++r) {
          int mm = mt * 16 + hi8 + r, nn = nt * 16 + lx;
          s_Am[mm * 32 + nn] = (bf16)((nn <= mm) ? acc[r] : 0.f);
        }
      }
    __syncthreads();

    // ---- o = A @ v + q_i @ S  (32x64) -> global ----
#pragma unroll
    for (int mt = 0; mt < 2; ++mt)
#pragma unroll
      for (int nt = 0; nt < 4; ++nt) {
        v8f acc = zero8();
        {
          v16bf a = frag_a(s_Am, mt * 16, 0, 32, lane);
          v16bf b = frag_bc(s_vT, 0, nt * 16, 32, lane);  // (k,n)->s_vT[n][k]
          acc = wmma_bf16(a, b, acc);
        }
#pragma unroll
        for (int kc = 0; kc < 2; ++kc) {
          v16bf a = frag_a(s_qi, mt * 16, kc * 32, 64, lane);
          v16bf b = frag_bc(s_SbT, kc * 32, nt * 16, 64, lane);
          acc = wmma_bf16(a, b, acc);
        }
#pragma unroll
        for (int r = 0; r < 8; ++r) {
          int mm = mt * 16 + hi8 + r;
          oG[(rb + mm) * 256 + colb + nt * 16 + lx] = acc[r];
        }
      }

    // ---- S = e^{gl[dk]} * S + k_o^T @ v ----
#pragma unroll
    for (int mt = 0; mt < 4; ++mt)
#pragma unroll
      for (int r = 0; r < 8; ++r) {
        float fgl = expf(s_gc[31 * 64 + mt * 16 + hi8 + r]);
#pragma unroll
        for (int nt = 0; nt < 4; ++nt) S[mt][nt][r] *= fgl;
      }
#pragma unroll
    for (int mt = 0; mt < 4; ++mt) {
      v16bf a = frag_a(s_koT, mt * 16, 0, 32, lane);  // (k_o^T)[dim][pos]
#pragma unroll
      for (int nt = 0; nt < 4; ++nt) {
        v16bf b = frag_bc(s_vT, 0, nt * 16, 32, lane);
        S[mt][nt] = wmma_bf16(a, b, S[mt][nt]);
      }
    }
    __syncthreads();
  }
}

// ---------------------------------------------------------------------------
// Per-head RMS norm * gn, SiLU(r) gate, pack bf16 for the Wo GEMM.
// ---------------------------------------------------------------------------
__global__ void gate_kernel(const float* __restrict__ oG,
                            const float* __restrict__ rG,
                            const float* __restrict__ gn,
                            bf16* __restrict__ gated, long Mrows) {
  long idx = blockIdx.x * (long)blockDim.x + threadIdx.x;
  if (idx >= Mrows * 4) return;
  int h = (int)(idx & 3);
  long row = idx >> 2;
  const float* po = oG + row * 256 + h * 64;
  float ms = 0.f;
  for (int d = 0; d < 64; ++d) { float v = po[d]; ms += v * v; }
  float inv = rsqrtf(ms / 64.f + LN_EPS);
  const float* pr = rG + row * 256 + h * 64;
  bf16* pg = gated + row * 256 + h * 64;
  for (int d = 0; d < 64; ++d) {
    float rv = pr[d];
    float sil = rv / (1.f + expf(-rv));
    pg[d] = (bf16)(po[d] * inv * gn[h * 64 + d] * sil);
  }
}

// Deconv as GEMM: dA[(n*Lp + pos), 2c+k] = g[n, pos-k, c] (0 out of range)
__global__ void deconv_pack_kernel(const float* __restrict__ g,
                                   bf16* __restrict__ dA, int Lp, int L,
                                   long total) {
  long idx = blockIdx.x * (long)blockDim.x + threadIdx.x;
  if (idx >= total) return;
  int col = (int)(idx & 255);
  long rp = idx >> 8;
  int pos = (int)(rp % Lp);
  int n   = (int)(rp / Lp);
  int c = col >> 1, k = col & 1;
  int l = pos - k;
  float val = (l >= 0 && l < L) ? g[((size_t)n * Lp + l) * 128 + c] : 0.f;
  dA[idx] = (bf16)val;
}

// Repack ct_w (D=128, Cout=64, K=2) -> W2[2c+k][o]
__global__ void w2_pack_kernel(const float* __restrict__ ctw,
                               float* __restrict__ W2) {
  int idx = blockIdx.x * blockDim.x + threadIdx.x;
  if (idx >= 256 * 64) return;
  int o = idx & 63, col = idx >> 6;
  int c = col >> 1, k = col & 1;
  W2[col * 64 + o] = ctw[c * 128 + o * 2 + k];
}

// dst[b,c,t,q] = base[b,c,t,q] + y2[row(b,t/q-major), c] + bias[c]
__global__ void residual_kernel(const float* __restrict__ base,
                                const float* __restrict__ y2,
                                const float* __restrict__ bias,
                                float* __restrict__ dst, int mode, long total) {
  long idx = blockIdx.x * (long)blockDim.x + threadIdx.x;
  if (idx >= total) return;
  int q = (int)(idx & 127);
  long r = idx >> 7;
  int t = (int)(r & 255); r >>= 8;
  int c = (int)(r & 63);
  int b = (int)(r >> 6);
  long row = (mode == 0) ? (((long)b * 256 + t) * 128 + q)
                         : (((long)b * 128 + q) * 256 + t);
  dst[idx] = base[idx] + y2[row * 64 + c] + bias[c];
}

// ---------------------------------------------------------------------------
extern "C" void kernel_launch(void* const* d_in, const int* in_sizes, int n_in,
                              void* d_out, int out_size, void* d_ws,
                              size_t ws_size, hipStream_t stream) {
  (void)in_sizes; (void)n_in; (void)out_size; (void)ws_size;
  const float* x = (const float*)d_in[0];  // (4,64,256,128); mouth unused

  struct Params {
    const float *gamma, *beta, *Wq, *Wk, *Wv, *Wg1, *Wg2, *Wr, *gn, *Wo, *ctw, *ctb;
  } pp[2];
  for (int p = 0; p < 2; ++p) {
    int b0 = 2 + p * 12;
    pp[p].gamma = (const float*)d_in[b0 + 0];
    pp[p].beta  = (const float*)d_in[b0 + 1];
    pp[p].Wq    = (const float*)d_in[b0 + 2];
    pp[p].Wk    = (const float*)d_in[b0 + 3];
    pp[p].Wv    = (const float*)d_in[b0 + 4];
    pp[p].Wg1   = (const float*)d_in[b0 + 5];
    pp[p].Wg2   = (const float*)d_in[b0 + 6];
    pp[p].Wr    = (const float*)d_in[b0 + 7];
    pp[p].gn    = (const float*)d_in[b0 + 8];
    pp[p].Wo    = (const float*)d_in[b0 + 9];
    pp[p].ctw   = (const float*)d_in[b0 + 10];
    pp[p].ctb   = (const float*)d_in[b0 + 11];
  }

  const long M    = 131072;   // Nn * Lp, identical for both passes
  const long BCTQ = 8388608;  // 4*64*256*128

  // ---- workspace carve (bump allocator, 256B aligned) ----
  char* wsb = (char*)d_ws;
  size_t off = 0;
  auto carve = [&](size_t bytes) -> void* {
    void* pr = wsb + off;
    off += (bytes + 255) & ~(size_t)255;
    return pr;
  };
  float* u     = (float*)carve((size_t)BCTQ * 4);
  bf16*  fA    = (bf16*) carve((size_t)M * 128 * 2);
  float* qB    = (float*)carve((size_t)M * 256 * 4);
  float* kB    = (float*)carve((size_t)M * 256 * 4);
  float* vB    = (float*)carve((size_t)M * 256 * 4);
  float* rB    = (float*)carve((size_t)M * 256 * 4);
  float* gkB   = (float*)carve((size_t)M * 256 * 4);
  float* oB    = (float*)carve((size_t)M * 256 * 4);
  bf16*  g1b   = (bf16*) carve((size_t)M * 32 * 2);
  bf16*  gated = (bf16*) carve((size_t)M * 256 * 2);
  float* gOut  = (float*)carve((size_t)M * 128 * 4);
  bf16*  dA    = (bf16*) carve((size_t)M * 256 * 2);
  float* y2    = (float*)carve((size_t)M * 64 * 4);
  float* W2    = (float*)carve((size_t)256 * 64 * 4);
  float* yBuf  = (float*)carve((size_t)BCTQ * 4);

  auto run_pass = [&](int p, const float* src, const float* rbase, float* dst) {
    const Params& w = pp[p];
    const int Lp = (p == 0) ? 128 : 256;
    const int L  = Lp - 1;
    const int nc = Lp / 32;
    const int Nn = (p == 0) ? 1024 : 512;  // B*T or B*Q
    const int inner = (p == 0) ? 256 : 128;
    const int strI  = (p == 0) ? 128 : 1;
    const int strP  = (p == 0) ? 1 : 128;

    // 1. LayerNorm over channels
    ln4d_kernel<<<(131072 + 255) / 256, 256, 0, stream>>>(
        src, w.gamma, w.beta, u, 64, 32768, 131072);

    // 2. Unfold -> bf16 A matrix [M x 128]
    long totF = M * 128;
    unfold_kernel<<<(unsigned)((totF + 255) / 256), 256, 0, stream>>>(
        u, fA, Lp, L, inner, 2097152, 32768, strI, strP, totF);

    // 3. Projections (WMMA GEMMs, K=128 -> N=256); q pre-scaled by Dk^-1/2
    dim3 gP(8, 2048);
    gemm_wmma_kernel<false><<<gP, 128, 0, stream>>>(fA, w.Wq, qB, nullptr, (int)M, 256, 128, 0.125f);
    gemm_wmma_kernel<false><<<gP, 128, 0, stream>>>(fA, w.Wk, kB, nullptr, (int)M, 256, 128, 1.f);
    gemm_wmma_kernel<false><<<gP, 128, 0, stream>>>(fA, w.Wv, vB, nullptr, (int)M, 256, 128, 1.f);
    gemm_wmma_kernel<false><<<gP, 128, 0, stream>>>(fA, w.Wr, rB, nullptr, (int)M, 256, 128, 1.f);

    // 4. Gate MLP: 128 -> 32 (bf16 out) -> 256, then log-sigmoid/32
    gemm_wmma_kernel<true><<<dim3(1, 2048), 128, 0, stream>>>(fA, w.Wg1, nullptr, g1b, (int)M, 32, 128, 1.f);
    gemm_wmma_kernel<false><<<gP, 128, 0, stream>>>(g1b, w.Wg2, gkB, nullptr, (int)M, 256, 32, 1.f);
    long totG = M * 256;
    logsig_kernel<<<(unsigned)((totG + 255) / 256), 256, 0, stream>>>(gkB, totG);

    // 5. Chunked GLA (intra-chunk WMMA attention + register-resident scan)
    gla_kernel<<<Nn * 4, 32, 0, stream>>>(qB, kB, vB, gkB, oB, nc);

    // 6. RMS norm + SiLU(r) gate -> bf16
    gate_kernel<<<(unsigned)((M * 4 + 255) / 256), 256, 0, stream>>>(oB, rB, w.gn, gated, M);

    // 7. Wo projection: 256 -> 128
    gemm_wmma_kernel<false><<<dim3(4, 2048), 128, 0, stream>>>(gated, w.Wo, gOut, nullptr, (int)M, 128, 256, 1.f);

    // 8. Transposed conv as GEMM: pack shifted features, 256 -> 64
    deconv_pack_kernel<<<(unsigned)((totG + 255) / 256), 256, 0, stream>>>(gOut, dA, Lp, L, totG);
    w2_pack_kernel<<<64, 256, 0, stream>>>(w.ctw, W2);
    gemm_wmma_kernel<false><<<dim3(2, 2048), 128, 0, stream>>>(dA, W2, y2, nullptr, (int)M, 64, 256, 1.f);

    // 9. Residual add back into (B,C,T,Q)
    residual_kernel<<<(unsigned)((BCTQ + 255) / 256), 256, 0, stream>>>(
        rbase, y2, w.ctb, dst, p, BCTQ);
  };

  run_pass(0, x, x, yBuf);                 // intra path (sequence along Q)
  run_pass(1, yBuf, yBuf, (float*)d_out);  // inter path (sequence along T)
}
